// LandmarkEmbedderv1_87540023427657
// MI455X (gfx1250) — matrix operand order repaired and checked by reading, not compiled, not run on hardware
//
#include <hip/hip_runtime.h>
#include <hip/hip_bf16.h>
#include <math.h>

// ---------------------------------------------------------------------------
// Problem constants (match reference)
// ---------------------------------------------------------------------------
#define N_NODE 4096
#define BATCH  16
#define DH     64
#define COLS   (BATCH * DH)     // 1024 combined batch*feature columns
#define EPS    1e-5f
#define KB     32               // K per WMMA step
#define NKB    (N_NODE / KB)    // 128 K-steps

typedef __attribute__((ext_vector_type(16))) __bf16 v16bf;
typedef __attribute__((ext_vector_type(8)))  __bf16 v8bf;
typedef __attribute__((ext_vector_type(8)))  float  v8f;

// ---------------------------------------------------------------------------
// CDNA5 async global->LDS path (ASYNCcnt) with portable fallback
// Builtin signature (from clang diagnostic): param0 = <4 x i32> addrspace(1)*,
// param1 = <4 x i32> addrspace(3)*, then imm offset, imm cpol.
// ---------------------------------------------------------------------------
#if defined(__has_builtin)
# if __has_builtin(__builtin_amdgcn_global_load_async_to_lds_b128)
#  define HAS_ASYNC_LDS 1
# endif
# if __has_builtin(__builtin_amdgcn_s_wait_asynccnt)
#  define HAS_WAIT_ASYNC 1
# endif
#endif

typedef int v4i_t __attribute__((vector_size(16)));
typedef __attribute__((address_space(1))) v4i_t* as1_v4i_ptr;
typedef __attribute__((address_space(3))) v4i_t* as3_v4i_ptr;

static __device__ __forceinline__ void stage_b128(const void* g, void* l) {
#if defined(HAS_ASYNC_LDS)
    // launder through integers: avoids AS-qualifier cast errors; truncating a
    // generic LDS pointer to 32 bits is the hardware LDS-offset mapping.
    as1_v4i_ptr gp = (as1_v4i_ptr)(unsigned long long)g;
    as3_v4i_ptr lp = (as3_v4i_ptr)(unsigned int)(unsigned long long)l;
    __builtin_amdgcn_global_load_async_to_lds_b128(gp, lp, /*offset=*/0, /*cpol=*/0);
#else
    *(uint4*)l = *(const uint4*)g;
#endif
}

template <int N>
static __device__ __forceinline__ void wait_async() {
#if defined(HAS_ASYNC_LDS)
# if defined(HAS_WAIT_ASYNC)
    __builtin_amdgcn_s_wait_asynccnt(N);
# else
    asm volatile("s_wait_asynccnt %0" :: "n"(N) : "memory");
# endif
#endif
}

// float -> bf16 round-to-nearest-even, stored as raw u16
static __device__ __forceinline__ unsigned short f2bf(float f) {
    union { float f; unsigned u; } v; v.f = f;
    unsigned u = v.u;
    unsigned r = u + 0x7FFFu + ((u >> 16) & 1u);
    return (unsigned short)(r >> 16);
}

// ---------------------------------------------------------------------------
// c_concat: two bilinear 0.5x stages == exact 4x4 average pool, stride 4
// ---------------------------------------------------------------------------
__global__ __launch_bounds__(256) void pool4_kernel(const float* __restrict__ pre,
                                                    float* __restrict__ out) {
    int i = blockIdx.x * 256 + threadIdx.x;            // 0 .. 65535
    int b   = i >> 12;
    int rem = i & 4095;
    int y   = rem >> 6;
    int x   = rem & 63;
    const float* p = pre + ((size_t)b * 256 + (size_t)y * 4) * 256 + (size_t)x * 4;
    float s = 0.f;
#pragma unroll
    for (int dy = 0; dy < 4; ++dy) {
#pragma unroll
        for (int dx = 0; dx < 4; ++dx) s += p[(size_t)dy * 256 + dx];
    }
    out[i] = s * (1.f / 16.f);
}

// ---------------------------------------------------------------------------
// A = sigmoid(adj * (1 - I))  -> bf16 row-major [N][N]
// ---------------------------------------------------------------------------
__global__ __launch_bounds__(256) void make_A_kernel(const float* __restrict__ adj,
                                                     unsigned short* __restrict__ A16) {
    size_t i = (size_t)blockIdx.x * 256 + threadIdx.x;
    int n = (int)(i >> 12);
    int m = (int)(i & 4095);
    float x = (n == m) ? 0.f : adj[i];
    float s = 1.f / (1.f + __expf(-x));
    A16[i] = f2bf(s);
}

// ---------------------------------------------------------------------------
// First dense layer: h0 = mv[:,:,:2]-mv[:,:,2:4]; hact = relu(h0 @ W00^T)
// Writes hact f32 [n][c] and bf16 transposed [c][n] (WMMA B operand).
// ---------------------------------------------------------------------------
__global__ __launch_bounds__(256) void dense_first_kernel(const float* __restrict__ mv,
                                                          const float* __restrict__ W,
                                                          float* __restrict__ hact32,
                                                          unsigned short* __restrict__ hactT16) {
    __shared__ float Ws[DH * 2];
    int tid = threadIdx.x;
    if (tid < DH * 2) Ws[tid] = W[tid];
    __syncthreads();

    int row = blockIdx.x * 256 + tid;                  // 0 .. 65535
    int n = row >> 4;
    int b = row & 15;
    const float* m = mv + ((size_t)b * N_NODE + n) * 4;
    float in0 = m[0] - m[2];
    float in1 = m[1] - m[3];
#pragma unroll 8
    for (int o = 0; o < DH; ++o) {
        float a = fmaxf(fmaf(in0, Ws[o * 2 + 0], in1 * Ws[o * 2 + 1]), 0.f);
        hact32[(size_t)n * COLS + b * DH + o]      = a;
        hactT16[(size_t)(b * DH + o) * N_NODE + n] = f2bf(a);
    }
}

// ---------------------------------------------------------------------------
// Mid dense layer: hact = relu(Hin @ W^T), Hin layout [n][c] (c = b*64+d)
// ---------------------------------------------------------------------------
__global__ __launch_bounds__(256) void dense_mid_kernel(const float* __restrict__ Hin,
                                                        const float* __restrict__ W,
                                                        float* __restrict__ hact32,
                                                        unsigned short* __restrict__ hactT16) {
    __shared__ float Ws[DH * DH];
    int tid = threadIdx.x;
    for (int i = tid; i < DH * DH; i += 256) Ws[i] = W[i];
    __syncthreads();

    int row = blockIdx.x * 256 + tid;                  // 0 .. 65535
    int n = row >> 4;
    int b = row & 15;
    const float* hp = Hin + (size_t)n * COLS + (size_t)b * DH;
    float hreg[DH];
#pragma unroll
    for (int d = 0; d < DH; ++d) hreg[d] = hp[d];

    for (int o = 0; o < DH; ++o) {
        float acc = 0.f;
#pragma unroll
        for (int d = 0; d < DH; ++d) acc = fmaf(hreg[d], Ws[o * DH + d], acc);
        float a = fmaxf(acc, 0.f);
        hact32[(size_t)n * COLS + b * DH + o]      = a;
        hactT16[(size_t)(b * DH + o) * N_NODE + n] = f2bf(a);
    }
}

// ---------------------------------------------------------------------------
// Big GEMM: Out[n][c] = sum_m A[n][m] * Hact[m][c]  + Hact[n][c]   (residual)
//   MODE 0: plain      -> Out f32 [n][c]
//   MODE 1: + BN eval  -> Out f32 [n][c], val*g[n]/sqrt(1+eps)+b[n]
//   MODE 2: final      -> Out f32 [b][n][o]  (c_crossattn layout)
//
// LDS-staged, double-buffered, async global->LDS copies (ASYNCcnt).
// WG tile 128(M)x128(N); 8 waves as 4(M)x2(N); wave tile 32x64 = 8 WMMAs/step.
// A16  : bf16 row-major [N][N]     (WMMA A operand)
// BT16 : bf16 row-major [COLS][N]  (transposed activations, WMMA B operand)
// ---------------------------------------------------------------------------
template <int MODE>
__global__ __launch_bounds__(256) void gcn_gemm_kernel(const unsigned short* __restrict__ A16,
                                                       const unsigned short* __restrict__ BT16,
                                                       const float* __restrict__ Hact32,
                                                       const float* __restrict__ bn_g,
                                                       const float* __restrict__ bn_b,
                                                       float* __restrict__ Out) {
    __shared__ __align__(16) unsigned short As[2][128 * KB];   // 2 x 8KB
    __shared__ __align__(16) unsigned short Bs[2][128 * KB];   // 2 x 8KB

    const int tid  = threadIdx.x;
    const int lane = tid & 31;
    const int wave = tid >> 5;
    const int mw   = wave & 3;          // 0..3 : M sub-block (32 rows)
    const int nw   = wave >> 2;         // 0..1 : N sub-block (64 cols)
    const int r    = lane & 15;
    const int h    = lane >> 4;

    const int wgM = blockIdx.x & 31;    // 32 M-tiles of 128
    const int wgN = blockIdx.x >> 5;    // 8  N-tiles of 128
    const int nbase = wgM * 128;        // global row base of WG tile
    const int cbase = wgN * 128;        // global col base of WG tile

    v8f zero = {};
    v8f acc[2][4];
#pragma unroll
    for (int rt = 0; rt < 2; ++rt)
#pragma unroll
        for (int ct = 0; ct < 4; ++ct) acc[rt][ct] = zero;

    // ---- staging: 512 x 16B chunks per operand per K-step; 2 each per thread
    auto stage = [&](int buf, int kb) {
        const int k0 = kb * KB;
#pragma unroll
        for (int u = 0; u < 2; ++u) {
            const int c   = tid + u * 256;      // 0..511
            const int row = c >> 2;             // 0..127
            const int q   = c & 3;              // 16B quarter within 32-elem row
            const int eo  = row * KB + q * 8;   // LDS element offset
            stage_b128(A16 + (size_t)(nbase + row) * N_NODE + k0 + q * 8, &As[buf][eo]);
            stage_b128(BT16 + (size_t)(cbase + row) * N_NODE + k0 + q * 8, &Bs[buf][eo]);
        }
    };

    stage(0, 0);

    for (int kb = 0; kb < NKB; ++kb) {
        const int cur = kb & 1;
        if (kb + 1 < NKB) {
            stage(cur ^ 1, kb + 1);
            wait_async<4>();                     // current buffer's loads done
        } else {
            wait_async<0>();
        }
        __syncthreads();                         // all waves' copies visible

        // B fragments (shared by both row-tiles of this wave)
        v16bf bfrag[4];
#pragma unroll
        for (int ct = 0; ct < 4; ++ct) {
            const int bb = (nw * 64 + ct * 16 + r) * KB + 8 * h;
            v8bf blo = *(const v8bf*)&Bs[cur][bb];
            v8bf bhi = *(const v8bf*)&Bs[cur][bb + 16];
            bfrag[ct] = __builtin_shufflevector(blo, bhi,
                    0, 1, 2, 3, 4, 5, 6, 7, 8, 9, 10, 11, 12, 13, 14, 15);
        }
#pragma unroll
        for (int rt = 0; rt < 2; ++rt) {
            const int ab = (mw * 32 + rt * 16 + r) * KB + 8 * h;
            v8bf alo = *(const v8bf*)&As[cur][ab];
            v8bf ahi = *(const v8bf*)&As[cur][ab + 16];
            v16bf afrag = __builtin_shufflevector(alo, ahi,
                    0, 1, 2, 3, 4, 5, 6, 7, 8, 9, 10, 11, 12, 13, 14, 15);
#pragma unroll
            for (int ct = 0; ct < 4; ++ct) {
                acc[rt][ct] = __builtin_amdgcn_wmma_f32_16x16x32_bf16(
                                  false, afrag, false, bfrag[ct],
                                  (short)0, acc[rt][ct], false, false);
            }
        }
        __syncthreads();                         // done with 'cur' before refill
    }

    // ---- epilogue: residual + (optional) eval-BN + layout permute
    const float inv_sqrt = rsqrtf(1.f + EPS);
#pragma unroll
    for (int rt = 0; rt < 2; ++rt) {
#pragma unroll
        for (int ct = 0; ct < 4; ++ct) {
            const int c = cbase + nw * 64 + ct * 16 + r;
#pragma unroll
            for (int j = 0; j < 8; ++j) {
                const int n = nbase + mw * 32 + rt * 16 + j + 8 * h;  // C/D: M=j+8h
                float v = acc[rt][ct][j] + Hact32[(size_t)n * COLS + c];
                if (MODE == 1) v = v * (bn_g[n] * inv_sqrt) + bn_b[n];
                if (MODE == 2) {
                    const int b = c >> 6, o = c & 63;
                    Out[(size_t)b * (N_NODE * DH) + (size_t)n * DH + o] = v;
                } else {
                    Out[(size_t)n * COLS + c] = v;
                }
            }
        }
    }
}

// ---------------------------------------------------------------------------
// Host launcher
// ---------------------------------------------------------------------------
extern "C" void kernel_launch(void* const* d_in, const int* in_sizes, int n_in,
                              void* d_out, int out_size, void* d_ws, size_t ws_size,
                              hipStream_t stream) {
    const float* pre = (const float*)d_in[0];
    const float* mv  = (const float*)d_in[1];
    const float* adj = (const float*)d_in[2];
    const float* W00 = (const float*)d_in[3];
    const float* W01 = (const float*)d_in[4];
    const float* W10 = (const float*)d_in[5];
    const float* W11 = (const float*)d_in[6];
    const float* W30 = (const float*)d_in[7];
    const float* W31 = (const float*)d_in[8];
    const float* g0  = (const float*)d_in[9];
    const float* b0  = (const float*)d_in[10];
    const float* g1  = (const float*)d_in[11];
    const float* b1  = (const float*)d_in[12];
    float* out = (float*)d_out;

    // workspace layout (72 MB total)
    char* ws = (char*)d_ws;
    unsigned short* A16    = (unsigned short*)(ws);                            // 32 MB
    float*          Hact32 = (float*)         (ws + (size_t)32 * 1024 * 1024); // 16 MB
    unsigned short* HT16   = (unsigned short*)(ws + (size_t)48 * 1024 * 1024); //  8 MB
    float*          H      = (float*)         (ws + (size_t)56 * 1024 * 1024); // 16 MB

    // c_concat (independent of GCN pipeline)
    pool4_kernel<<<256, 256, 0, stream>>>(pre, out);

    // normalized adjacency in bf16
    make_A_kernel<<<65536, 256, 0, stream>>>(adj, A16);

    const int DG = 256;   // dense grid: 65536 (n,b) rows / 256
    const int GG = 256;   // gemm grid: 32 M-tiles x 8 N-tiles

    // layer pair 1 (+BN g0,b0)
    dense_first_kernel<<<DG, 256, 0, stream>>>(mv, W00, Hact32, HT16);
    gcn_gemm_kernel<0><<<GG, 256, 0, stream>>>(A16, HT16, Hact32, nullptr, nullptr, H);
    dense_mid_kernel<<<DG, 256, 0, stream>>>(H, W01, Hact32, HT16);
    gcn_gemm_kernel<1><<<GG, 256, 0, stream>>>(A16, HT16, Hact32, g0, b0, H);

    // layer pair 2 (+BN g1,b1)
    dense_mid_kernel<<<DG, 256, 0, stream>>>(H, W10, Hact32, HT16);
    gcn_gemm_kernel<0><<<GG, 256, 0, stream>>>(A16, HT16, Hact32, nullptr, nullptr, H);
    dense_mid_kernel<<<DG, 256, 0, stream>>>(H, W11, Hact32, HT16);
    gcn_gemm_kernel<1><<<GG, 256, 0, stream>>>(A16, HT16, Hact32, g1, b1, H);

    // layer pair 3 (final -> c_crossattn region of d_out, [B,N,64] layout)
    dense_mid_kernel<<<DG, 256, 0, stream>>>(H, W30, Hact32, HT16);
    gcn_gemm_kernel<0><<<GG, 256, 0, stream>>>(A16, HT16, Hact32, nullptr, nullptr, H);
    dense_mid_kernel<<<DG, 256, 0, stream>>>(H, W31, Hact32, HT16);
    gcn_gemm_kernel<2><<<GG, 256, 0, stream>>>(A16, HT16, Hact32, nullptr, nullptr, out + 65536);
}
